// Controller_60816736911410
// MI455X (gfx1250) — matrix operand order repaired and checked by reading, not compiled, not run on hardware
//
#include <hip/hip_runtime.h>
#include <hip/hip_bf16.h>
#include <math.h>

typedef __attribute__((ext_vector_type(16))) _Float16 v16h;
typedef __attribute__((ext_vector_type(8)))  float    v8f;

#define N_AGENTS 4096
#define TOPK 32

// Branchless helpers on packed 64-bit ordered keys.
__device__ __forceinline__ unsigned long long umin64(unsigned long long a,
                                                     unsigned long long b) {
    return (b < a) ? b : a;   // v_cmp_lt_u64 + v_cndmask (or v_min_u64)
}
__device__ __forceinline__ unsigned long long umax64(unsigned long long a,
                                                     unsigned long long b) {
    return (b > a) ? b : a;
}

// ---------------------------------------------------------------------------
// Kernel A: per-agent 32-nearest-neighbor selection.
// Packed key = (d2_bits << 32) | j  -> unsigned-min gives argmin with exact
// lowest-index tie-break (jax.lax.top_k stability). One block per agent.
// ---------------------------------------------------------------------------
__global__ __launch_bounds__(256) void topk_kernel(const float* __restrict__ states,
                                                   int* __restrict__ idxWs,
                                                   float* __restrict__ maskWs) {
    __shared__ float d2[N_AGENTS];
    __shared__ unsigned long long wk[8];
    const int t = threadIdx.x, agent = blockIdx.x;
    const int lane = t & 31, w = t >> 5;

    const float sx = states[agent * 4 + 0];
    const float sy = states[agent * 4 + 1];
    for (int j = t; j < N_AGENTS; j += 256) {
        float dx = sx - states[j * 4 + 0];
        float dy = sy - states[j * 4 + 1];
        d2[j] = dx * dx + dy * dy;       // sqrt(+1e-4) is monotone in d2
    }
    __syncthreads();

    for (int r = 0; r < TOPK; ++r) {
        unsigned long long bk = ~0ull;
        for (int j = t; j < N_AGENTS; j += 256) {
            unsigned long long k =
                ((unsigned long long)__float_as_uint(d2[j]) << 32) | (unsigned)j;
            bk = umin64(bk, k);
        }
        // wave-level reduce (branchless)
#pragma unroll
        for (int m = 1; m <= 16; m <<= 1)
            bk = umin64(bk, (unsigned long long)__shfl_xor((unsigned long long)bk, m, 32));
        if (lane == 0) wk[w] = bk;
        __syncthreads();
        if (t == 0) {
            unsigned long long fk = wk[0];
#pragma unroll
            for (int q = 1; q < 8; ++q) fk = umin64(fk, wk[q]);
            int   fi = (int)(fk & 0xffffffffu);
            float fv = __uint_as_float((unsigned)(fk >> 32));
            idxWs[agent * TOPK + r]  = fi;
            maskWs[agent * TOPK + r] = (fv < 1.0f) ? 1.0f : 0.0f;   // dist<1 <=> d2<1
            d2[fi] = 3.4e38f;
        }
        __syncthreads();
    }
}

// ---------------------------------------------------------------------------
// Kernel B: fused conv stack + masked argmax.
// Block = 256 threads = 8 waves, 32 agents; wave w owns M-tile mt=w of H2.
//   phase 1: build H1 = relu(W1@Hreshape + b1) in f16 into LDS [pos][ch].
//   phase 2: H2 tile = W2_tile @ H1 via v_wmma_f32_16x16x32_f16 (K=64 -> 2
//            k-steps, N=32 -> 2 n-tiles), bias+relu+mask, packed-key argmax.
// ---------------------------------------------------------------------------
__global__ __launch_bounds__(256) void conv_argmax_kernel(const float* __restrict__ states,
                                                          const float* __restrict__ W1,
                                                          const float* __restrict__ b1,
                                                          const float* __restrict__ W2,
                                                          const float* __restrict__ b2,
                                                          const int* __restrict__ idxWs,
                                                          const float* __restrict__ maskWs,
                                                          float* __restrict__ xloc) {
    __shared__ alignas(32) _Float16 sH1[32 * 64];   // [pos][ch]
    __shared__ float sMask[32];

    const int t    = threadIdx.x;
    const int lane = t & 31;
    const int mt   = t >> 5;                 // wave id == M-tile id (0..7)
    const int colL = lane & 15;              // within-tile column
    const int hiA  = (lane < 16) ? 0 : 16;   // A-fragment half offset
    const int hiB  = (lane < 16) ? 0 : 16;   // B-fragment half offset

    // Hoisted A fragments: W2 tile rows m = mt*16 + colL, f32 -> f16.
    // A k order chosen here (16x32 f16, contiguous): lane<16 halves h -> K = h; lane>=16: +16.
    const int m = mt * 16 + colL;
    v16h Af0, Af1;
#pragma unroll
    for (int h = 0; h < 16; ++h) {
        int kk = h + hiA;
        Af0[h] = (_Float16)W2[m * 64 + kk];        // k-step 0
        Af1[h] = (_Float16)W2[m * 64 + 32 + kk];   // k-step 1
    }
    // Hoisted biases: C layout row = mt*16 + r + (lane>=16 ? 8 : 0)
    float bias[8];
#pragma unroll
    for (int r = 0; r < 8; ++r) bias[r] = b2[mt * 16 + r + ((lane >= 16) ? 8 : 0)];

    const int p = t & 31;               // position handled in phase 1
    const int cbase = (t >> 5) * 8;     // 8 channels per thread

    for (int al = 0; al < 32; ++al) {
        const int agent = blockIdx.x * 32 + al;

        // ---- phase 1: build H1 for this agent --------------------------------
        const float sa0 = states[agent * 4 + 0], sa1 = states[agent * 4 + 1];
        const float sa2 = states[agent * 4 + 2], sa3 = states[agent * 4 + 3];
        // raw reshape [32,5]->[5,32]: h[c][p] = xk[(c*32+p)/5][(c*32+p)%5]
        float hcol[5];
#pragma unroll
        for (int c = 0; c < 5; ++c) {
            int flat = c * 32 + p;
            int k = flat / 5, e = flat - 5 * k;
            int j = idxWs[agent * TOPK + k];
            float se = (e == 0) ? sa0 : (e == 1) ? sa1 : (e == 2) ? sa2 : sa3;
            float diff = se - states[j * 4 + (e & 3)];
            float self = (j == agent) ? 1.0f : 0.0f;
            hcol[c] = (e == 4) ? self : diff;
        }
#pragma unroll
        for (int cc = 0; cc < 8; ++cc) {
            int c = cbase + cc;
            float acc = b1[c];
#pragma unroll
            for (int e = 0; e < 5; ++e) acc += W1[c * 5 + e] * hcol[e];
            sH1[p * 64 + c] = (_Float16)fmaxf(acc, 0.0f);
        }
        if (t < 32) sMask[t] = maskWs[agent * TOPK + t];
        __syncthreads();

        // ---- phase 2: WMMA H2 tile + masked argmax ---------------------------
        // B k order chosen here (contiguous, matches A): lane column n = colL, halves h -> K = hiB + h,
        // contiguous in sH1 (channel-contiguous per position), 32B aligned.
        v8f acc0 = {}, acc1 = {};
        {
            v16h B;
            B = *(const v16h*)(sH1 + (colL)      * 64 +  0 + hiB);   // nt0 ks0
            acc0 = __builtin_amdgcn_wmma_f32_16x16x32_f16(false, Af0, false, B,
                                                          (short)0, acc0, false, false);
            B = *(const v16h*)(sH1 + (colL)      * 64 + 32 + hiB);   // nt0 ks1
            acc0 = __builtin_amdgcn_wmma_f32_16x16x32_f16(false, Af1, false, B,
                                                          (short)0, acc0, false, false);
            B = *(const v16h*)(sH1 + (16 + colL) * 64 +  0 + hiB);   // nt1 ks0
            acc1 = __builtin_amdgcn_wmma_f32_16x16x32_f16(false, Af0, false, B,
                                                          (short)0, acc1, false, false);
            B = *(const v16h*)(sH1 + (16 + colL) * 64 + 32 + hiB);   // nt1 ks1
            acc1 = __builtin_amdgcn_wmma_f32_16x16x32_f16(false, Af1, false, B,
                                                          (short)0, acc1, false, false);
        }
        const float m0 = sMask[colL];
        const float m1 = sMask[16 + colL];
        // Fully unrolled: direct register refs (no m0/v_movrels indirection) and
        // 8 independent bpermute/max chains the scheduler can interleave.
#pragma unroll
        for (int r = 0; r < 8; ++r) {
            float v0 = fmaxf(acc0[r] + bias[r], 0.0f) * m0;   // col = colL
            float v1 = fmaxf(acc1[r] + bias[r], 0.0f) * m1;   // col = 16+colL
            // Packed key = (val_bits << 32) | (31 - col): unsigned-max gives
            // argmax with first-occurrence (lowest col) tie-break; val >= 0.
            unsigned long long k0 =
                ((unsigned long long)__float_as_uint(v0) << 32) | (unsigned)(31 - colL);
            unsigned long long k1 =
                ((unsigned long long)__float_as_uint(v1) << 32) | (unsigned)(15 - colL);
            unsigned long long bk = umax64(k0, k1);
#pragma unroll
            for (int mm = 1; mm <= 8; mm <<= 1)
                bk = umax64(bk, (unsigned long long)__shfl_xor((unsigned long long)bk, mm, 32));
            if (colL == 0) {
                int row = mt * 16 + r + ((lane >= 16) ? 8 : 0);
                xloc[agent * 128 + row] = (float)(31 - (int)(bk & 31u));
            }
        }
        __syncthreads();
    }
}

// ---------------------------------------------------------------------------
// Kernel C: per-agent MLP 132->64->128->64->4, gains, final accelerations.
// One wave per agent (8 agents per 256-thread block), LDS ping-pong buffers.
// ---------------------------------------------------------------------------
__global__ __launch_bounds__(256) void mlp_kernel(const float* __restrict__ states,
                                                  const float* __restrict__ goals,
                                                  const float* __restrict__ Wd1,
                                                  const float* __restrict__ bd1,
                                                  const float* __restrict__ Wd2,
                                                  const float* __restrict__ bd2,
                                                  const float* __restrict__ Wd3,
                                                  const float* __restrict__ bd3,
                                                  const float* __restrict__ Wd4,
                                                  const float* __restrict__ bd4,
                                                  const float* __restrict__ xloc,
                                                  float* __restrict__ out) {
    __shared__ float bufA[8][132];
    __shared__ float bufB[8][128];
    const int lane = threadIdx.x & 31, w = threadIdx.x >> 5;
    const int agent = blockIdx.x * 8 + w;
    float* A = bufA[w];
    float* B = bufB[w];

    // feat = [x_local(128), rel(2), vel(2)]
#pragma unroll
    for (int o = lane; o < 128; o += 32) A[o] = xloc[agent * 128 + o];
    if (lane == 0) {
        A[128] = states[agent * 4 + 0] - goals[agent * 2 + 0];
        A[129] = states[agent * 4 + 1] - goals[agent * 2 + 1];
        A[130] = states[agent * 4 + 2];
        A[131] = states[agent * 4 + 3];
    }
    __syncthreads();
    // L1: 64 <- 132
    for (int o = lane; o < 64; o += 32) {
        float acc = bd1[o];
        for (int k = 0; k < 132; ++k) acc += Wd1[o * 132 + k] * A[k];
        B[o] = fmaxf(acc, 0.0f);
    }
    __syncthreads();
    // L2: 128 <- 64
    for (int o = lane; o < 128; o += 32) {
        float acc = bd2[o];
        for (int k = 0; k < 64; ++k) acc += Wd2[o * 64 + k] * B[k];
        A[o] = fmaxf(acc, 0.0f);
    }
    __syncthreads();
    // L3: 64 <- 128
    for (int o = lane; o < 64; o += 32) {
        float acc = bd3[o];
        for (int k = 0; k < 128; ++k) acc += Wd3[o * 128 + k] * A[k];
        B[o] = fmaxf(acc, 0.0f);
    }
    __syncthreads();
    // L4: 4 <- 64, gains k = 2*sigmoid(z)+0.2
    if (lane < 4) {
        float acc = bd4[lane];
        for (int k = 0; k < 64; ++k) acc += Wd4[lane * 64 + k] * B[k];
        A[lane] = 2.0f / (1.0f + expf(-acc)) + 0.2f;
    }
    __syncthreads();
    if (lane == 0) {
        float rx = states[agent * 4 + 0] - goals[agent * 2 + 0];
        float ry = states[agent * 4 + 1] - goals[agent * 2 + 1];
        float vx = states[agent * 4 + 2];
        float vy = states[agent * 4 + 3];
        out[agent * 2 + 0] = -(A[0] * rx + A[1] * vx);
        out[agent * 2 + 1] = -(A[2] * ry + A[3] * vy);
    }
}

// ---------------------------------------------------------------------------
extern "C" void kernel_launch(void* const* d_in, const int* in_sizes, int n_in,
                              void* d_out, int out_size, void* d_ws, size_t ws_size,
                              hipStream_t stream) {
    (void)in_sizes; (void)n_in; (void)out_size; (void)ws_size;
    const float* states = (const float*)d_in[0];
    const float* goals  = (const float*)d_in[1];
    const float* W1     = (const float*)d_in[2];
    const float* b1     = (const float*)d_in[3];
    const float* W2     = (const float*)d_in[4];
    const float* b2     = (const float*)d_in[5];
    const float* Wd1    = (const float*)d_in[6];
    const float* bd1    = (const float*)d_in[7];
    const float* Wd2    = (const float*)d_in[8];
    const float* bd2    = (const float*)d_in[9];
    const float* Wd3    = (const float*)d_in[10];
    const float* bd3    = (const float*)d_in[11];
    const float* Wd4    = (const float*)d_in[12];
    const float* bd4    = (const float*)d_in[13];
    float* out = (float*)d_out;

    // workspace: idx (512KB int) | mask (512KB f32) | x_local (2MB f32)
    char* ws = (char*)d_ws;
    int*   idxWs  = (int*)ws;
    float* maskWs = (float*)(ws + (size_t)N_AGENTS * TOPK * sizeof(int));
    float* xloc   = (float*)(ws + 2 * (size_t)N_AGENTS * TOPK * sizeof(int));

    topk_kernel<<<N_AGENTS, 256, 0, stream>>>(states, idxWs, maskWs);
    conv_argmax_kernel<<<N_AGENTS / 32, 256, 0, stream>>>(states, W1, b1, W2, b2,
                                                          idxWs, maskWs, xloc);
    mlp_kernel<<<N_AGENTS / 8, 256, 0, stream>>>(states, goals, Wd1, bd1, Wd2, bd2,
                                                 Wd3, bd3, Wd4, bd4, xloc, out);
}
